// DecoderRNN_75548474736873
// MI455X (gfx1250) — matrix operand order repaired and checked by reading, not compiled, not run on hardware
//
#include <hip/hip_runtime.h>

// ---------------- constants ----------------
static constexpr int Hh   = 1024;   // hidden
static constexpr int Ee   = 512;    // embedding
static constexpr int Vv   = 32000;  // vocab
static constexpr int Tt   = 256;    // target length
static constexpr int H3   = 3072;   // 3*H
static constexpr int BOS  = 2;

// ---------------- types ----------------
typedef __bf16 v8bf  __attribute__((ext_vector_type(8)));
typedef __bf16 v16bf __attribute__((ext_vector_type(16)));
typedef float  v8f   __attribute__((ext_vector_type(8)));

// ---------------- helpers ----------------
__device__ __forceinline__ unsigned short f2bf(float f) {
    unsigned u = __builtin_bit_cast(unsigned, f);
    u += 0x7fffu + ((u >> 16) & 1u);           // round-to-nearest-even
    return (unsigned short)(u >> 16);
}
__device__ __forceinline__ float bflo(unsigned u) {
    return __builtin_bit_cast(float, u << 16);
}
__device__ __forceinline__ float bfhi(unsigned u) {
    return __builtin_bit_cast(float, u & 0xffff0000u);
}
__device__ __forceinline__ float wave_sum(float v) {
#pragma unroll
    for (int o = 16; o > 0; o >>= 1) v += __shfl_xor(v, o, 32);
    return v;
}
__device__ __forceinline__ float wave_max(float v) {
#pragma unroll
    for (int o = 16; o > 0; o >>= 1) v = fmaxf(v, __shfl_xor(v, o, 32));
    return v;
}
__device__ __forceinline__ float sigm(float x) { return 1.0f / (1.0f + __expf(-x)); }

// ---------------- init: h0 = relu(W_w @ ctx + W_b) ----------------
// grid 128, block 256 (8 waves; 1 row per wave)
__global__ void k_h0init(const float* __restrict__ W, const float* __restrict__ b,
                         const float* __restrict__ ctx, float* __restrict__ h0) {
    __shared__ float cs[2 * Hh];
    for (int i = threadIdx.x; i < 2 * Hh; i += 256) cs[i] = ctx[i];
    __syncthreads();
    const int wave = threadIdx.x >> 5, lane = threadIdx.x & 31;
    const int row  = blockIdx.x * 8 + wave;
    const float* wr = W + (size_t)row * (2 * Hh);
    float acc = 0.f;
    for (int k = lane; k < 2 * Hh; k += 32) acc += wr[k] * cs[k];
    acc = wave_sum(acc);
    if (lane == 0) h0[row] = fmaxf(acc + b[row], 0.f);
}

// ---------------- embedding gather + relu -> bf16, plus h1 init ----------------
// grid Tt + 4, block 256
__global__ void k_embed(const float* __restrict__ emb, const int* __restrict__ y,
                        const float* __restrict__ wemb,
                        unsigned short* __restrict__ xs_bf, float* __restrict__ h1) {
    const int bx = blockIdx.x;
    if (bx < Tt) {
        const int tok = (bx == 0) ? BOS : y[bx - 1];
        const float* src = emb + (size_t)tok * Ee;
        unsigned short* dst = xs_bf + (size_t)bx * Ee;
        for (int e = threadIdx.x; e < Ee; e += 256)
            dst[e] = f2bf(fmaxf(src[e], 0.f));
    } else {
        const int j = (bx - Tt) * 256 + threadIdx.x;
        h1[j] = wemb[j];
    }
}

// ---------------- f32 -> bf16 conversion (grid-stride) ----------------
__global__ void k_cvt_bf16(const float* __restrict__ src, unsigned short* __restrict__ dst,
                           long long n) {
    long long i = (long long)blockIdx.x * blockDim.x + threadIdx.x;
    const long long stride = (long long)gridDim.x * blockDim.x;
    for (; i < n; i += stride) dst[i] = f2bf(src[i]);
}

// ---------------- WMMA bf16 GEMM: C[M x N] = A[MxK] * B[NxK]^T + bias ----------------
// Block 256 thr (8 waves). Block tile = 128 rows x 128 cols.
// Each wave: 8 M-tiles x 1 N-tile (128x16). A strip staged in LDS (padded),
// B fragments streamed from global (read once per 128 output rows).
// Per K-step: gather all 8 A-fragments into distinct registers, then issue
// 8 back-to-back WMMAs (avoids per-fragment dscnt stalls).
// grid.x = N/128, grid.y = M/128
template <int K>
__global__ void k_wmma_gemm(const unsigned short* __restrict__ A,
                            const unsigned short* __restrict__ B,
                            const float* __restrict__ bias,
                            float* __restrict__ C, int N) {
    constexpr int APAD = 40;  // padded row stride in halves (conflict-free fragment reads)
    __shared__ __align__(16) unsigned short As[128 * APAD];

    const int tid  = threadIdx.x;
    const int lane = tid & 31;
    const int wave = tid >> 5;
    const int h    = lane >> 4;     // lane half
    const int l15  = lane & 15;
    const int row0 = blockIdx.y * 128;
    const int col0 = blockIdx.x * 128 + wave * 16;

    const unsigned short* bRow = B + (size_t)(col0 + l15) * K;  // B fragment: N = l15

    v8f acc[8];
#pragma unroll
    for (int mt = 0; mt < 8; ++mt) acc[mt] = v8f{0.f, 0.f, 0.f, 0.f, 0.f, 0.f, 0.f, 0.f};

    for (int k0 = 0; k0 < K; k0 += 32) {
        // stage A[128 x 32] chunk into LDS: 512 chunks of 8 halves, 2 per thread
#pragma unroll
        for (int it = 0; it < 2; ++it) {
            const int id = tid + it * 256;
            const int r  = id >> 2;
            const int c  = id & 3;
            *reinterpret_cast<v8bf*>(&As[r * APAD + c * 8]) =
                *reinterpret_cast<const v8bf*>(A + (size_t)(row0 + r) * K + k0 + c * 8);
        }
        __syncthreads();

        // B 32x16 bf16 fragment: elems 0..7 -> K=k0+8h+e ; 8..15 -> K=k0+16+8h+(e-8)
        v8bf b_lo = *reinterpret_cast<const v8bf*>(bRow + k0 + 8 * h);
        v8bf b_hi = *reinterpret_cast<const v8bf*>(bRow + k0 + 16 + 8 * h);
        v16bf bf = __builtin_shufflevector(b_lo, b_hi, 0, 1, 2, 3, 4, 5, 6, 7,
                                           8, 9, 10, 11, 12, 13, 14, 15);

        // gather all 8 A fragments first (one ds clause, one wait) ...
        v16bf af[8];
#pragma unroll
        for (int mt = 0; mt < 8; ++mt) {
            // A 16x32 fragment: M = l15; elems 0..7 -> K=8h+e ; 8..15 -> K=16+8h+(e-8)
            const unsigned short* ar = &As[(mt * 16 + l15) * APAD];
            v8bf a_lo = *reinterpret_cast<const v8bf*>(ar + 8 * h);
            v8bf a_hi = *reinterpret_cast<const v8bf*>(ar + 16 + 8 * h);
            af[mt] = __builtin_shufflevector(a_lo, a_hi, 0, 1, 2, 3, 4, 5, 6, 7,
                                             8, 9, 10, 11, 12, 13, 14, 15);
        }
        // ... then issue the WMMA burst back-to-back
#pragma unroll
        for (int mt = 0; mt < 8; ++mt) {
            acc[mt] = __builtin_amdgcn_wmma_f32_16x16x32_bf16(false, af[mt], false, bf,
                                                              (short)0, acc[mt], false, false);
        }
        __syncthreads();
    }

    const int n  = col0 + l15;
    const float bn = bias[n];
#pragma unroll
    for (int mt = 0; mt < 8; ++mt) {
#pragma unroll
        for (int r = 0; r < 8; ++r) {
            const int m = row0 + mt * 16 + r + 8 * h;   // D: VGPR r, lane-half selects M block
            C[(size_t)m * N + n] = acc[mt][r] + bn;
        }
    }
}

// ---------------- scan: gh0 = Wh0bf @ h0(t) + bh0 ----------------
// grid 96, block 1024 (32 waves, 1 row/wave)
__global__ void k_gh0(const unsigned short* __restrict__ Wbf, const float* __restrict__ bh,
                      const float* __restrict__ h, float* __restrict__ gh) {
    __shared__ float hs[Hh];
    hs[threadIdx.x] = h[threadIdx.x];
    __syncthreads();
    const int wave = threadIdx.x >> 5, lane = threadIdx.x & 31;
    const int row  = blockIdx.x * 32 + wave;
    const unsigned* wr = reinterpret_cast<const unsigned*>(Wbf + (size_t)row * Hh);
    float acc = 0.f;
#pragma unroll
    for (int i = 0; i < Hh / 64; ++i) {
        const unsigned u = wr[i * 32 + lane];
        const int k = (i * 32 + lane) * 2;
        acc += bflo(u) * hs[k] + bfhi(u) * hs[k + 1];
    }
    acc = wave_sum(acc);
    if (lane == 0) gh[row] = acc + bh[row];
}

// ---------------- fused: h0(t+1)+h1(t) updates, then gi1/gh1 dots ----------------
// grid 96, block 1024 (32 waves, 1 row/wave)
// Each block redundantly computes both GRU gate updates into LDS; block 0 also
// persists the new states (and the previous output row) to global.
__global__ void k_g1_fused(const unsigned short* __restrict__ Wi_bf,
                           const unsigned short* __restrict__ Wh_bf,
                           const float* __restrict__ bi, const float* __restrict__ bh,
                           const float* __restrict__ gi0_t,    // gi0 row t
                           const float* __restrict__ gh0,      // gh0(t)
                           const float* __restrict__ h0_prev,  // h0(t)
                           float* __restrict__ h0_out,         // h0(t+1)
                           const float* __restrict__ gi1_prev, // step t-1 gates
                           const float* __restrict__ gh1_prev,
                           const float* __restrict__ h1_prev,  // h1(t-1)
                           float* __restrict__ h1_out,         // h1(t)
                           unsigned short* __restrict__ out_bf_prev, // outs row t-1
                           float* __restrict__ gi1_out, float* __restrict__ gh1_out,
                           int first) {
    __shared__ float h0s[Hh];
    __shared__ float h1s[Hh];
    const int j = threadIdx.x;  // 1024 threads, one hidden unit each

    // layer-0 update: h0(t+1)
    {
        const float r = sigm(gi0_t[j] + gh0[j]);
        const float z = sigm(gi0_t[Hh + j] + gh0[Hh + j]);
        const float n = tanhf(gi0_t[2 * Hh + j] + r * gh0[2 * Hh + j]);
        h0s[j] = (1.f - z) * n + z * h0_prev[j];
    }
    // layer-1 update: h1(t) from previous step's gates
    float h1v;
    if (first) {
        h1v = h1_prev[j];
    } else {
        const float r = sigm(gi1_prev[j] + gh1_prev[j]);
        const float z = sigm(gi1_prev[Hh + j] + gh1_prev[Hh + j]);
        const float n = tanhf(gi1_prev[2 * Hh + j] + r * gh1_prev[2 * Hh + j]);
        h1v = (1.f - z) * n + z * h1_prev[j];
    }
    h1s[j] = h1v;
    if (blockIdx.x == 0) {
        h0_out[j] = h0s[j];
        h1_out[j] = h1v;
        if (!first) out_bf_prev[j] = f2bf(h1v);
    }
    __syncthreads();

    // dots: gi1 = Wi1 @ h0(t+1) + bi1 ; gh1 = Wh1 @ h1(t) + bh1
    const int wave = threadIdx.x >> 5, lane = threadIdx.x & 31;
    const int row  = blockIdx.x * 32 + wave;
    const unsigned* wi = reinterpret_cast<const unsigned*>(Wi_bf + (size_t)row * Hh);
    const unsigned* wh = reinterpret_cast<const unsigned*>(Wh_bf + (size_t)row * Hh);
    float ai = 0.f, ah = 0.f;
#pragma unroll
    for (int i = 0; i < Hh / 64; ++i) {
        const int idx = i * 32 + lane;
        const int k   = idx * 2;
        const unsigned ui = wi[idx];
        const unsigned uh = wh[idx];
        ai += bflo(ui) * h0s[k] + bfhi(ui) * h0s[k + 1];
        ah += bflo(uh) * h1s[k] + bfhi(uh) * h1s[k + 1];
    }
    ai = wave_sum(ai);
    ah = wave_sum(ah);
    if (lane == 0) { gi1_out[row] = ai + bi[row]; gh1_out[row] = ah + bh[row]; }
}

// ---------------- tail: final layer-1 update + emit last output row ----------------
__global__ void k_upd1(const float* __restrict__ gi, const float* __restrict__ gh,
                       const float* __restrict__ hprev, float* __restrict__ hnext,
                       unsigned short* __restrict__ out_bf) {
    const int j = threadIdx.x;
    const float r = sigm(gi[j] + gh[j]);
    const float z = sigm(gi[Hh + j] + gh[Hh + j]);
    const float n = tanhf(gi[2 * Hh + j] + r * gh[2 * Hh + j]);
    const float hn = (1.f - z) * n + z * hprev[j];
    hnext[j]  = hn;
    out_bf[j] = f2bf(hn);
}

// ---------------- in-place log_softmax over Vv per row ---------------- (grid Tt, block 1024)
__global__ void k_logsoftmax(float* __restrict__ x) {
    __shared__ float red[32];
    float* row = x + (size_t)blockIdx.x * Vv;
    const int tid = threadIdx.x;

    float mx = -3.402823466e38f;
    for (int i = tid; i < Vv; i += 1024) mx = fmaxf(mx, row[i]);
    mx = wave_max(mx);
    if ((tid & 31) == 0) red[tid >> 5] = mx;
    __syncthreads();
    mx = wave_max(red[tid & 31]);          // all 32 partials reduced redundantly per wave
    __syncthreads();

    float s = 0.f;
    for (int i = tid; i < Vv; i += 1024) s += __expf(row[i] - mx);
    s = wave_sum(s);
    if ((tid & 31) == 0) red[tid >> 5] = s;
    __syncthreads();
    s = wave_sum(red[tid & 31]);

    const float lse = mx + __logf(s);
    for (int i = tid; i < Vv; i += 1024) row[i] -= lse;
}

// ---------------- host launcher ----------------
extern "C" void kernel_launch(void* const* d_in, const int* in_sizes, int n_in,
                              void* d_out, int out_size, void* d_ws, size_t ws_size,
                              hipStream_t stream) {
    const float* word_embedding = (const float*)d_in[0];
    const float* context_vector = (const float*)d_in[1];
    const int*   y              = (const int*)d_in[2];
    // d_in[3] = target_length (constant Tt) unused
    const float* W_w  = (const float*)d_in[4];
    const float* W_b  = (const float*)d_in[5];
    const float* emb  = (const float*)d_in[6];
    const float* Wi0  = (const float*)d_in[7];
    const float* Wh0  = (const float*)d_in[8];
    const float* bi0  = (const float*)d_in[9];
    const float* bh0  = (const float*)d_in[10];
    const float* Wi1  = (const float*)d_in[11];
    const float* Wh1  = (const float*)d_in[12];
    const float* bi1  = (const float*)d_in[13];
    const float* bh1  = (const float*)d_in[14];
    const float* cls_W = (const float*)d_in[15];
    const float* cls_b = (const float*)d_in[16];
    float* out = (float*)d_out;

    // -------- workspace carve --------
    char* p = (char*)d_ws;
    auto alloc = [&](size_t bytes) -> char* {
        char* r = p;
        p += (bytes + 255) & ~(size_t)255;
        return r;
    };
    unsigned short* xs_bf   = (unsigned short*)alloc((size_t)Tt * Ee * 2);
    unsigned short* Wi0_bf  = (unsigned short*)alloc((size_t)H3 * Ee * 2);
    unsigned short* Wh0_bf  = (unsigned short*)alloc((size_t)H3 * Hh * 2);
    unsigned short* Wi1_bf  = (unsigned short*)alloc((size_t)H3 * Hh * 2);
    unsigned short* Wh1_bf  = (unsigned short*)alloc((size_t)H3 * Hh * 2);
    unsigned short* clsW_bf = (unsigned short*)alloc((size_t)Vv * Hh * 2);
    unsigned short* outs_bf = (unsigned short*)alloc((size_t)Tt * Hh * 2);
    float* gi0  = (float*)alloc((size_t)Tt * H3 * 4);
    float* gh0  = (float*)alloc((size_t)H3 * 4);
    float* gi1b = (float*)alloc(2 * (size_t)H3 * 4);
    float* gh1b = (float*)alloc(2 * (size_t)H3 * 4);
    float* h0b  = (float*)alloc(2 * (size_t)Hh * 4);
    float* h1b  = (float*)alloc(2 * (size_t)Hh * 4);
    float* gi1buf[2] = {gi1b, gi1b + H3};
    float* gh1buf[2] = {gh1b, gh1b + H3};
    float* h0buf[2]  = {h0b, h0b + Hh};
    float* h1buf[2]  = {h1b, h1b + Hh};

    // -------- init --------
    k_h0init<<<dim3(Hh / 8), dim3(256), 0, stream>>>(W_w, W_b, context_vector, h0buf[0]);
    k_embed<<<dim3(Tt + 4), dim3(256), 0, stream>>>(emb, y, word_embedding, xs_bf, h1buf[0]);

    // -------- weight conversions to bf16 (L2-resident working set) --------
    k_cvt_bf16<<<dim3(2048), dim3(256), 0, stream>>>(Wi0, Wi0_bf, (long long)H3 * Ee);
    k_cvt_bf16<<<dim3(4096), dim3(256), 0, stream>>>(Wh0, Wh0_bf, (long long)H3 * Hh);
    k_cvt_bf16<<<dim3(4096), dim3(256), 0, stream>>>(Wi1, Wi1_bf, (long long)H3 * Hh);
    k_cvt_bf16<<<dim3(4096), dim3(256), 0, stream>>>(Wh1, Wh1_bf, (long long)H3 * Hh);
    k_cvt_bf16<<<dim3(8192), dim3(256), 0, stream>>>(cls_W, clsW_bf, (long long)Vv * Hh);

    // -------- gi0 for all timesteps: [Tt x H3] = xs[Tt x Ee] * Wi0^T + bi0 (WMMA) --------
    k_wmma_gemm<Ee><<<dim3(H3 / 128, Tt / 128), dim3(256), 0, stream>>>(
        xs_bf, Wi0_bf, bi0, gi0, H3);

    // -------- sequential GRU scan: 2 launches per step --------
    for (int t = 0; t < Tt; ++t) {
        const int cur = t & 1, nxt = cur ^ 1;
        k_gh0<<<dim3(H3 / 32), dim3(1024), 0, stream>>>(Wh0_bf, bh0, h0buf[cur], gh0);
        k_g1_fused<<<dim3(H3 / 32), dim3(1024), 0, stream>>>(
            Wi1_bf, Wh1_bf, bi1, bh1,
            gi0 + (size_t)t * H3, gh0, h0buf[cur], h0buf[nxt],
            gi1buf[nxt], gh1buf[nxt],            // step t-1 gates (parity (t-1)&1 == nxt)
            h1buf[cur], h1buf[nxt],
            outs_bf + (size_t)(t > 0 ? t - 1 : 0) * Hh,
            gi1buf[cur], gh1buf[cur],            // step t gates written to parity t&1
            t == 0 ? 1 : 0);
    }
    // tail: emit h1(256) gates -> outs row 255   (h1(255) is in h1buf[(255+1)&1] = h1buf[0])
    k_upd1<<<dim3(1), dim3(Hh), 0, stream>>>(gi1buf[1], gh1buf[1], h1buf[0], h1buf[1],
                                             outs_bf + (size_t)(Tt - 1) * Hh);

    // -------- classifier GEMM straight into d_out: logits = outs * cls_W^T + cls_b --------
    k_wmma_gemm<Hh><<<dim3(Vv / 128, Tt / 128), dim3(256), 0, stream>>>(
        outs_bf, clsW_bf, cls_b, out, Vv);

    // -------- in-place log_softmax --------
    k_logsoftmax<<<dim3(Tt), dim3(1024), 0, stream>>>(out);
}
